// KPlanes_8229157339535
// MI455X (gfx1250) — matrix operand order repaired and checked by reading, not compiled, not run on hardware
//
#include <hip/hip_runtime.h>
#include <hip/hip_bf16.h>

// ---------------------------------------------------------------------------
// K-Planes field eval on gfx1250 (CDNA5, wave32, WMMA).
// Per wave: 32 points = two 16-row M-tiles sharing every B-operand fetch.
// Plane gathers: planes pre-transposed to channel-last f16 (HWC) in workspace
// so each bilinear corner is two contiguous 16B loads in exact WMMA-A layout.
// MLP: chained v_wmma_f32_16x16x32_f16 with f32 accumulation; inter-layer
// lane-transpose via per-wave LDS rows; ReLU via v_med3 (1 VALU/elem).
// ---------------------------------------------------------------------------

typedef __attribute__((ext_vector_type(16))) _Float16 v16h;
typedef __attribute__((ext_vector_type(8)))  _Float16 v8h;
typedef __attribute__((ext_vector_type(8)))  float    v8f;

union V16 { v16h v; v8h h8[2]; _Float16 e[16]; };

static __device__ __forceinline__ v8f wmma16x16x32(v16h a, v16h b, v8f c) {
  // 8 args: (neg_a, A, neg_b, B, c_mod, C, reuse_a, reuse_b)
  return __builtin_amdgcn_wmma_f32_16x16x32_f16(false, a, false, b, (short)0, c,
                                                false, false);
}

static __device__ __forceinline__ void lds_fence() {
  asm volatile("s_wait_dscnt 0" ::: "memory");
}

static __device__ __forceinline__ float relu(float x) {
#if __has_builtin(__builtin_amdgcn_fmed3f)
  return __builtin_amdgcn_fmed3f(x, 0.0f, __builtin_inff());  // v_med3 relu
#else
  return fmaxf(x, 0.0f);
#endif
}

// Workspace layout (f16 halves).
// Weights: tile (kt,nt) = 512 halves = one full B register set.
#define OFF_FW1 0        // 32x64   : Kt=1  Nt=4  ->  2048
#define OFF_FW2 2048     // 64x64   : Kt=2  Nt=4  ->  4096
#define OFF_W1  6144     // 119x256 : Kt=4  Nt=16 -> 32768 (K zero-padded to 128)
#define OFF_W2  38912    // 256x256 : Kt=8  Nt=16 -> 65536
#define OFF_W3  104448   // 256x256 : Kt=8  Nt=16 -> 65536
#define OFF_W4  169984   // 256x3   : Kt=8  Nt=1  ->  4096 (N zero-padded to 16)
#define OFF_PYX 174080                       // yx: 540*960*32 halves
#define OFF_PXT (OFF_PYX + 540 * 960 * 32)   // xt: 960*300*32 halves
#define OFF_PYT (OFF_PXT + 960 * 300 * 32)   // yt: 540*300*32 halves
#define WS_HALVES_TOTAL (OFF_PYT + 540 * 300 * 32)  // 31,162,880 halves
#define WS_BYTES_FAST ((size_t)WS_HALVES_TOTAL * 2) // ~62.3 MB
#define WS_BYTES_MIN  ((size_t)OFF_PYX * 2)         // weights only

#define N_PTS 1048576
#define PTS_PER_WAVE 32   // two 16-row M-tiles

// ---------------------------------------------------------------------------
// Weight pre-swizzle: W (K x N, f32 row-major) -> f16 WMMA-B layout.
//   column n = nt*16 + lane%16 ; row k = kt*32 + (half_i/8)*16 + (lane/16)*8 + half_i%8
// ---------------------------------------------------------------------------
__global__ void kp_prep_weights(const float* __restrict__ W, _Float16* __restrict__ dst,
                                int K, int N, int Kt, int Nt) {
  int e = blockIdx.x * blockDim.x + threadIdx.x;
  int total = Kt * Nt * 512;
  if (e >= total) return;
  int i    = e & 15;
  int lane = (e >> 4) & 31;
  int tile = e >> 9;
  int nt   = tile % Nt;
  int kt   = tile / Nt;
  int n = nt * 16 + (lane & 15);
  int k = kt * 32 + ((i >> 3) << 4) + (lane >> 4) * 8 + (i & 7);
  float v = (k < K && n < N) ? W[k * N + n] : 0.0f;
  dst[e] = (_Float16)v;
}

// Plane transpose (f,H,W) f32 -> (H*W, 32) f16. Writes fully coalesced;
// reads are 32 parallel linear streams (one per channel plane).
__global__ void kp_prep_plane(const float* __restrict__ src,
                              _Float16* __restrict__ dst, int HW) {
  size_t e = (size_t)blockIdx.x * blockDim.x + threadIdx.x;
  size_t total = (size_t)HW * 32;
  if (e >= total) return;
  int c = (int)(e & 31);
  size_t pos = e >> 5;
  dst[e] = (_Float16)src[(size_t)c * HW + pos];
}

// ---------------------------------------------------------------------------
// Bilinear setup (align_corners=False, border padding); offsets in texels.
// ---------------------------------------------------------------------------
struct SampG {
  int o00, o01, o10, o11;
  float wx, wy;
};

static __device__ __forceinline__ SampG mk_samp(int H, int W, float a, float b) {
  float ix = fminf(fmaxf(a * (float)W - 0.5f, 0.0f), (float)(W - 1));
  float iy = fminf(fmaxf(b * (float)H - 0.5f, 0.0f), (float)(H - 1));
  float xf = floorf(ix), yf = floorf(iy);
  int x0 = (int)xf, y0 = (int)yf;
  int x1 = min(x0 + 1, W - 1);
  int y1 = min(y0 + 1, H - 1);
  SampG s;
  s.o00 = y0 * W + x0; s.o01 = y0 * W + x1;
  s.o10 = y1 * W + x0; s.o11 = y1 * W + x1;
  s.wx = ix - xf; s.wy = iy - yf;
  return s;
}

// Fast path: HWC f16 plane; 8 consecutive channels starting at cb.
static __device__ __forceinline__ void bil8(const _Float16* __restrict__ base,
                                            const SampG& s, int cb, float* r) {
  v8h q00 = *(const v8h*)(base + (size_t)s.o00 * 32 + cb);
  v8h q01 = *(const v8h*)(base + (size_t)s.o01 * 32 + cb);
  v8h q10 = *(const v8h*)(base + (size_t)s.o10 * 32 + cb);
  v8h q11 = *(const v8h*)(base + (size_t)s.o11 * 32 + cb);
#pragma unroll
  for (int i = 0; i < 8; ++i) {
    float v00 = (float)q00[i], v01 = (float)q01[i];
    float v10 = (float)q10[i], v11 = (float)q11[i];
    float t0 = v00 + (v01 - v00) * s.wx;
    float t1 = v10 + (v11 - v10) * s.wx;
    r[i] = t0 + (t1 - t0) * s.wy;
  }
}

// Fallback path: channel-major f32 plane, single channel.
static __device__ __forceinline__ float samp32(const float* __restrict__ base,
                                               const SampG& s, int c, int HW) {
  const float* b = base + (size_t)c * (size_t)HW;
  float v00 = b[s.o00], v01 = b[s.o01], v10 = b[s.o10], v11 = b[s.o11];
  float t0 = v00 + (v01 - v00) * s.wx;
  float t1 = v10 + (v11 - v10) * s.wx;
  return t0 + (t1 - t0) * s.wy;
}

// Load NKT A-tiles (16 x 32 each) for one 16-row M-tile from LDS rows.
template <int NKT>
static __device__ __forceinline__ void load_a(V16* a, const _Float16* L,
                                              int stride, int row, int g) {
#pragma unroll
  for (int kt = 0; kt < NKT; ++kt) {
    const _Float16* pr = L + row * stride + kt * 32 + g * 8;
    a[kt].h8[0] = *(const v8h*)pr;
    a[kt].h8[1] = *(const v8h*)(pr + 16);
  }
}

// One dense layer for two M-tiles sharing each B fetch: out = relu(in @ W + b)
template <int KT, int NT>
static __device__ __forceinline__ void layer2m(const V16* aA, const V16* aB,
                                               const _Float16* wsb,
                                               const float* __restrict__ bias,
                                               _Float16* outL, int outStride,
                                               int lane) {
  const int g = lane >> 4, m = lane & 15;
  for (int nt = 0; nt < NT; ++nt) {
    float bv = bias[nt * 16 + m];
    v8f acc0 = {bv, bv, bv, bv, bv, bv, bv, bv};
    v8f acc1 = acc0;
#pragma unroll
    for (int kt = 0; kt < KT; ++kt) {
      v16h b = *(const v16h*)(wsb + (kt * NT + nt) * 512 + lane * 16);
      acc0 = wmma16x16x32(aA[kt].v, b, acc0);
      acc1 = wmma16x16x32(aB[kt].v, b, acc1);
    }
#pragma unroll
    for (int r = 0; r < 8; ++r) {
      outL[(r + 8 * g) * outStride + nt * 16 + m] = (_Float16)relu(acc0[r]);
      outL[(16 + r + 8 * g) * outStride + nt * 16 + m] = (_Float16)relu(acc1[r]);
    }
  }
}

// ---------------------------------------------------------------------------
// Main kernel: block = 1 wave (32 threads), 32 points / wave.
// HWC=true reads f16 channel-last planes from workspace; else f32 gathers.
// ---------------------------------------------------------------------------
template <bool HWC>
__global__ __launch_bounds__(32) void kp_main(
    const float* __restrict__ xyt,
    const float* __restrict__ yx_plane,   // (32,540,960) f32
    const float* __restrict__ xt_plane,   // (32,960,300) f32
    const float* __restrict__ yt_plane,   // (32,540,300) f32
    const _Float16* __restrict__ pyx,     // (540*960,32) f16
    const _Float16* __restrict__ pxt,     // (960*300,32) f16
    const _Float16* __restrict__ pyt,     // (540*300,32) f16
    const float* __restrict__ fb1,
    const float* __restrict__ fb2,
    const float* __restrict__ b1,
    const float* __restrict__ b2,
    const float* __restrict__ b3,
    const float* __restrict__ b4,
    const int*   __restrict__ frames,
    const _Float16* __restrict__ ws,
    float* __restrict__ out) {

  // 40 KB per-wave LDS (8 waves/WGP against the 320 KB pool)
  __shared__ _Float16 s_full[32 * 128];  // [h2 | yx_pe | t_pe | pad]
  __shared__ _Float16 s_zA[32 * 256];
  __shared__ _Float16 s_zB[32 * 256];    // also overlays h1 (32 x 64)

  const int lane = threadIdx.x & 31;
  const int g    = lane >> 4;   // K-subgroup for A / row-group for D
  const int m    = lane & 15;   // point-in-tile for A, N-in-tile for D
  const int base = blockIdx.x * PTS_PER_WAVE;

  const float inv_fn = 1.0f / (float)frames[0];

  // --- plane sampling for both M-tiles (points base+m, base+16+m) ----------
  float xs[2], ys[2], ts[2];
  V16 a0[2];
#pragma unroll
  for (int mt = 0; mt < 2; ++mt) {
    const int p = base + mt * 16 + m;
    const float x = xyt[3 * p + 0];
    const float y = xyt[3 * p + 1];
    const float t = xyt[3 * p + 2] * inv_fn;
    xs[mt] = x; ys[mt] = y; ts[mt] = t;
    SampG s1 = mk_samp(540, 960, y, x);  // uv=[y,x]
    SampG s2 = mk_samp(960, 300, x, t);  // uv=[x,t]
    SampG s3 = mk_samp(540, 300, y, t);  // uv=[y,t]
    if constexpr (HWC) {
      // Lane's channel groups: jg=0 -> c in [g*8, g*8+8); jg=1 -> +16
#pragma unroll
      for (int jg = 0; jg < 2; ++jg) {
        const int cb = jg * 16 + g * 8;
        float r1[8], r2[8], r3[8];
        bil8(pyx, s1, cb, r1);
        bil8(pxt, s2, cb, r2);
        bil8(pyt, s3, cb, r3);
#pragma unroll
        for (int i = 0; i < 8; ++i)
          a0[mt].e[jg * 8 + i] = (_Float16)(r1[i] * r2[i] * r3[i]);
      }
    } else {
      // half j -> channel c = (j/8)*16 + (lane/16)*8 + (j%8): exact A layout
#pragma unroll
      for (int j = 0; j < 16; ++j) {
        int c = ((j >> 3) << 4) + (g << 3) + (j & 7);
        float f = samp32(yx_plane, s1, c, 540 * 960) *
                  samp32(xt_plane, s2, c, 960 * 300) *
                  samp32(yt_plane, s3, c, 540 * 300);
        a0[mt].e[j] = (_Float16)f;
      }
    }
  }

  // ---------------- GEMM1: feat(32x32) @ fw1(32x64) + fb1, relu -> h1 ------
  _Float16* hL = s_zB;  // 32 x 64, row stride 64 (overlays zB)
  layer2m<1, 4>(&a0[0], &a0[1], ws + OFF_FW1, fb1, hL, 64, lane);
  lds_fence();

  // ---------------- GEMM2: h1(32x64) @ fw2(64x64) + fb2, relu -> full[0:64]
  V16 aA[8], aB[8];
  load_a<2>(aA, hL, 64, m, g);
  load_a<2>(aB, hL, 64, 16 + m, g);
  layer2m<2, 4>(aA, aB, ws + OFF_FW2, fb2, s_full, 128, lane);

  // ---------------- positional encodings -> full[64:119], zero pad ---------
  {
    // lane l owns row l: l<16 -> tile-0 coords, l>=16 -> tile-1 coords
    const float px = (lane < 16) ? xs[0] : xs[1];
    const float py = (lane < 16) ? ys[0] : ys[1];
    const float pt = (lane < 16) ? ts[0] : ts[1];
    _Float16* row = s_full + lane * 128;
    row[64] = (_Float16)py;
    row[65] = (_Float16)px;
#pragma unroll
    for (int j = 0; j < 10; ++j) {
      float sc = (float)(1 << j);
      row[66 + 4 * j + 0] = (_Float16)__sinf(sc * py);
      row[66 + 4 * j + 1] = (_Float16)__sinf(sc * px);
      row[66 + 4 * j + 2] = (_Float16)__cosf(sc * py);
      row[66 + 4 * j + 3] = (_Float16)__cosf(sc * px);
    }
    row[106] = (_Float16)pt;
#pragma unroll
    for (int j = 0; j < 6; ++j) {
      float sc = (float)(1 << j);
      row[107 + 2 * j] = (_Float16)__sinf(sc * pt);
      row[108 + 2 * j] = (_Float16)__cosf(sc * pt);
    }
#pragma unroll
    for (int k = 119; k < 128; ++k) row[k] = (_Float16)0.0f;
  }
  lds_fence();

  // ---------------- GEMM3: full(32x128) @ w1(128x256) + b1, relu -> zA -----
  load_a<4>(aA, s_full, 128, m, g);
  load_a<4>(aB, s_full, 128, 16 + m, g);
  layer2m<4, 16>(aA, aB, ws + OFF_W1, b1, s_zA, 256, lane);
  lds_fence();

  // ---------------- GEMM4: zA @ w2 + b2, relu -> zB ------------------------
  load_a<8>(aA, s_zA, 256, m, g);
  load_a<8>(aB, s_zA, 256, 16 + m, g);
  layer2m<8, 16>(aA, aB, ws + OFF_W2, b2, s_zB, 256, lane);
  lds_fence();

  // ---------------- GEMM5: zB @ w3 + b3, relu -> zA ------------------------
  load_a<8>(aA, s_zB, 256, m, g);
  load_a<8>(aB, s_zB, 256, 16 + m, g);
  layer2m<8, 16>(aA, aB, ws + OFF_W3, b3, s_zA, 256, lane);
  lds_fence();

  // ---------------- GEMM6: zA @ w4(256x3 padded to 16) + b4, sigmoid -------
  {
    load_a<8>(aA, s_zA, 256, m, g);
    load_a<8>(aB, s_zA, 256, 16 + m, g);
    const _Float16* wsb = ws + OFF_W4;
    float bv = (m < 3) ? b4[m] : 0.0f;
    v8f acc0 = {bv, bv, bv, bv, bv, bv, bv, bv};
    v8f acc1 = acc0;
#pragma unroll
    for (int kt = 0; kt < 8; ++kt) {
      v16h b = *(const v16h*)(wsb + kt * 512 + lane * 16);
      acc0 = wmma16x16x32(aA[kt].v, b, acc0);
      acc1 = wmma16x16x32(aB[kt].v, b, acc1);
    }
    if (m < 3) {
#pragma unroll
      for (int r = 0; r < 8; ++r) {
        float s0 = 1.0f / (1.0f + __expf(-acc0[r]));
        float s1 = 1.0f / (1.0f + __expf(-acc1[r]));
        out[(size_t)(base + r + 8 * g) * 3 + m] = s0;
        out[(size_t)(base + 16 + r + 8 * g) * 3 + m] = s1;
      }
    }
  }
}

// ---------------------------------------------------------------------------
extern "C" void kernel_launch(void* const* d_in, const int* in_sizes, int n_in,
                              void* d_out, int out_size, void* d_ws, size_t ws_size,
                              hipStream_t stream) {
  const float* xyt = (const float*)d_in[0];
  const float* yxp = (const float*)d_in[1];
  const float* xtp = (const float*)d_in[2];
  const float* ytp = (const float*)d_in[3];
  const float* fw1 = (const float*)d_in[4];
  const float* fb1 = (const float*)d_in[5];
  const float* fw2 = (const float*)d_in[6];
  const float* fb2 = (const float*)d_in[7];
  const float* w1  = (const float*)d_in[8];
  const float* b1  = (const float*)d_in[9];
  const float* w2  = (const float*)d_in[10];
  const float* b2  = (const float*)d_in[11];
  const float* w3  = (const float*)d_in[12];
  const float* b3  = (const float*)d_in[13];
  const float* w4  = (const float*)d_in[14];
  const float* b4  = (const float*)d_in[15];
  const int*   fn  = (const int*)d_in[16];
  float* out = (float*)d_out;
  _Float16* ws = (_Float16*)d_ws;

  // Pre-swizzle all weights into WMMA-B f16 layout (idempotent, deterministic).
  struct P { const float* W; int off, K, N, Kt, Nt; };
  const P preps[6] = {
      {fw1, OFF_FW1,  32,  64, 1,  4},
      {fw2, OFF_FW2,  64,  64, 2,  4},
      {w1,  OFF_W1,  119, 256, 4, 16},
      {w2,  OFF_W2,  256, 256, 8, 16},
      {w3,  OFF_W3,  256, 256, 8, 16},
      {w4,  OFF_W4,  256,   3, 8,  1},
  };
  for (int i = 0; i < 6; ++i) {
    int total = preps[i].Kt * preps[i].Nt * 512;
    kp_prep_weights<<<(total + 255) / 256, 256, 0, stream>>>(
        preps[i].W, ws + preps[i].off, preps[i].K, preps[i].N, preps[i].Kt,
        preps[i].Nt);
  }

  const bool fast = ws_size >= WS_BYTES_FAST;
  if (fast) {
    // Transpose planes to channel-last f16 (idempotent each call).
    struct T { const float* src; int off, HW; };
    const T tps[3] = {
        {yxp, OFF_PYX, 540 * 960},
        {xtp, OFF_PXT, 960 * 300},
        {ytp, OFF_PYT, 540 * 300},
    };
    for (int i = 0; i < 3; ++i) {
      size_t total = (size_t)tps[i].HW * 32;
      kp_prep_plane<<<(unsigned)((total + 255) / 256), 256, 0, stream>>>(
          tps[i].src, ws + tps[i].off, tps[i].HW);
    }
  }

  const int blocks = N_PTS / PTS_PER_WAVE;  // 32768
  if (fast) {
    kp_main<true><<<blocks, 32, 0, stream>>>(
        xyt, yxp, xtp, ytp, ws + OFF_PYX, ws + OFF_PXT, ws + OFF_PYT,
        fb1, fb2, b1, b2, b3, b4, fn, ws, out);
  } else {
    kp_main<false><<<blocks, 32, 0, stream>>>(
        xyt, yxp, xtp, ytp, nullptr, nullptr, nullptr,
        fb1, fb2, b1, b2, b3, b4, fn, ws, out);
  }
}
